// PredictionAttention_42236708389351
// MI455X (gfx1250) — matrix operand order repaired and checked by reading, not compiled, not run on hardware
//
#include <hip/hip_runtime.h>
#include <hip/hip_bf16.h>
#include <math.h>

typedef __attribute__((ext_vector_type(16))) __bf16 v16bf;
typedef __attribute__((ext_vector_type(4)))  __bf16 v4bf;
typedef __attribute__((ext_vector_type(8)))  float  v8f;
typedef __attribute__((ext_vector_type(2)))  float  v2f;
typedef __bf16 bf16_t;

#define B_SZ 4096
#define C_SZ 32
#define T_SZ 9
#define L_SZ 24
#define NH   8
#define HD   4
#define TL   216   /* T*L rows */

/* ---- workspace offsets (bytes), written by prep_kernel ---- */
#define WS_QW   0        /* bf16 [32*32] q_w */
#define WS_KW   2048     /* bf16 [32*32] k_w */
#define WS_VMB  4096     /* bf16 [9][32][192]: (V @ m_w^T) fused, k contiguous */
#define WS_CONV 114688   /* bf16 [9][32][64]: [tap][oc][ic], ic contiguous */

/* ---- LDS layout (bytes); row-padded so all WMMA stores are unconditional ---- */
#define OFF_YT   0       /* bf16 [224][32] tar rows */
#define OFF_YC   14336   /* bf16 [224][32] con rows */
#define OFF_ZQ   28672   /* f32 [224][32] */
#define OFF_ZK   57344   /* f32 [224][32] */
#define OFF_S    86016   /* f32 [8][32][32] per-t raw scores */
#define OFF_PCAT 118784  /* bf16 [32][192] per-t probabilities, heads concatenated */
#define OFF_INP  131072  /* bf16 [224][64] conv input: ch0-31=con2, ch32-63=z_m */
#define OFF_ZROW 159744  /* bf16 [64] zero row for conv halo */
#define OFF_H    159872  /* f32 [224][33] */
#define OFF_TP   189440  /* f32 [32][9][6] */
#define OFF_QW   196352  /* bf16 [1024] */
#define OFF_KW   198400  /* bf16 [1024] */
#define OFF_PW   200448  /* f32 [144] */
#define OFF_LPW  201024  /* f32 [144] */
#define OFF_CB   201600  /* f32 [32] conv_b */
#define OFF_BG   201728  /* f32 [32] bn_g */
#define OFF_BB   201856  /* f32 [32] bn_b */
#define OFF_PB   201984  /* f32 [6] */
#define OFF_LPB  202008  /* f32 [24] */
#define SMEM_BYTES 202112

/* ===== prep: b-invariant v-head, fused V@m_w^T, bf16 weight repack ===== */
__global__ __launch_bounds__(256)
void prep_kernel(const float* __restrict__ rule_prompt,
                 const float* __restrict__ v_w, const float* __restrict__ v_b,
                 const float* __restrict__ q_w, const float* __restrict__ k_w,
                 const float* __restrict__ m_w, const float* __restrict__ conv_w,
                 char* __restrict__ ws) {
  __shared__ float z[TL * 32];
  const int tid = threadIdx.x;
  for (int i = tid; i < TL * 32; i += 256) {
    const int row = i >> 5, o = i & 31;
    float s = v_b[o];
    #pragma unroll
    for (int c = 0; c < 32; ++c) s += rule_prompt[c * TL + row] * v_w[o * 32 + c];
    z[i] = s;
  }
  __syncthreads();
  /* normalize each (row, head) group of 4 in place */
  for (int g = tid; g < TL * NH; g += 256) {
    float* pz = z + (g >> 3) * 32 + (g & 7) * 4;
    const float s = pz[0]*pz[0] + pz[1]*pz[1] + pz[2]*pz[2] + pz[3]*pz[3];
    const float nn = sqrtf(s);
    const float inv = 1.0f / fmaxf(nn, 1e-12f);
    pz[0]*=inv; pz[1]*=inv; pz[2]*=inv; pz[3]*=inv;
  }
  __syncthreads();
  /* VMB[t][o][h*24+lk] = sum_d vn[t*24+lk][h*4+d] * m_w[o][h*4+d] */
  bf16_t* vmb = (bf16_t*)(ws + WS_VMB);
  for (int i = tid; i < T_SZ * 32 * 192; i += 256) {
    const int t = i / (32 * 192), rem = i % (32 * 192);
    const int o = rem / 192, k = rem % 192;
    const int h = k / L_SZ, lk = k % L_SZ;
    const float* vz = z + (t * L_SZ + lk) * 32 + h * 4;
    const float* mw = m_w + o * 32 + h * 4;
    vmb[i] = (bf16_t)(vz[0]*mw[0] + vz[1]*mw[1] + vz[2]*mw[2] + vz[3]*mw[3]);
  }
  bf16_t* qw = (bf16_t*)(ws + WS_QW);
  bf16_t* kw = (bf16_t*)(ws + WS_KW);
  for (int i = tid; i < 1024; i += 256) { qw[i] = (bf16_t)q_w[i]; kw[i] = (bf16_t)k_w[i]; }
  bf16_t* wc = (bf16_t*)(ws + WS_CONV);
  for (int i = tid; i < 9 * 32 * 64; i += 256) {
    const int tap = i / 2048, rem = i % 2048;
    const int oc = rem >> 6, ic = rem & 63;
    const int dt = tap / 3, dl = tap % 3;
    wc[i] = (bf16_t)conv_w[((oc * 64 + ic) * 3 + dt) * 3 + dl];
  }
}

/* ======================= fused per-batch kernel ======================= */
__global__ __launch_bounds__(256, 1)
void fused_kernel(const float* __restrict__ x,
                  const float* __restrict__ pos_prompt,
                  const float* __restrict__ pre_prompt,
                  const float* __restrict__ q_b, const float* __restrict__ k_b,
                  const float* __restrict__ m_b,
                  const float* __restrict__ conv_b, const float* __restrict__ bn_g,
                  const float* __restrict__ bn_b,
                  const float* __restrict__ p_w, const float* __restrict__ p_b,
                  const float* __restrict__ lp_w, const float* __restrict__ lp_b,
                  const int* __restrict__ atten_flag,
                  const char* __restrict__ ws,
                  float* __restrict__ out) {
  extern __shared__ char smem[];
  bf16_t* yT  = (bf16_t*)(smem + OFF_YT);
  bf16_t* yC  = (bf16_t*)(smem + OFF_YC);
  float*  zq  = (float*)(smem + OFF_ZQ);
  float*  zk  = (float*)(smem + OFF_ZK);
  float*  S   = (float*)(smem + OFF_S);
  bf16_t* Pc  = (bf16_t*)(smem + OFF_PCAT);
  bf16_t* inp = (bf16_t*)(smem + OFF_INP);
  bf16_t* zrow= (bf16_t*)(smem + OFF_ZROW);
  float*  hL  = (float*)(smem + OFF_H);
  float*  tpL = (float*)(smem + OFF_TP);
  bf16_t* qwL = (bf16_t*)(smem + OFF_QW);
  bf16_t* kwL = (bf16_t*)(smem + OFF_KW);
  float*  pwL = (float*)(smem + OFF_PW);
  float*  lpwL= (float*)(smem + OFF_LPW);
  float*  cbL = (float*)(smem + OFF_CB);
  float*  bgL = (float*)(smem + OFF_BG);
  float*  bbL = (float*)(smem + OFF_BB);
  float*  pbL = (float*)(smem + OFF_PB);
  float*  lpbL= (float*)(smem + OFF_LPB);

  const int b    = blockIdx.x;
  const int tid  = threadIdx.x;
  const int lane = tid & 31;
  const int wv   = tid >> 5;
  const int flag = atten_flag[0];
  const int lo   = (flag == 1) ? 18 : (flag == 2) ? 12 : (flag == 3) ? 6 : 0;
  const float* xb = x + (size_t)b * (C_SZ * TL);

  /* ---- P1: splice tar/con/con2/tp; stage weights; zero pads ---- */
  for (int i = tid; i < C_SZ * TL; i += 256) {
    const int c = i / TL, pos = i % TL;
    const float xv = xb[i];
    const float pp = pos_prompt[i];
    const float pr = pre_prompt[i];
    const int l = pos % L_SZ;
    const bool inw = (l >= lo) && (l < lo + 6);
    yT[pos * 32 + c]  = (bf16_t)(inw ? xv : pp);   /* tar */
    yC[pos * 32 + c]  = (bf16_t)(inw ? pp : xv);   /* con */
    inp[pos * 64 + c] = (bf16_t)(inw ? pr : xv);   /* con2 */
    if (inw) tpL[(c * T_SZ + pos / L_SZ) * 6 + (l - lo)] = xv;
  }
  { /* zero pads: tar/con rows 216..223, conv zero-row, Pcat rows 24..31 */
    const int pos = TL + wv;
    yT[pos * 32 + lane] = (bf16_t)0.0f;
    yC[pos * 32 + lane] = (bf16_t)0.0f;
    if (tid < 64) zrow[tid] = (bf16_t)0.0f;
    for (int i = tid; i < 8 * 192; i += 256) Pc[24 * 192 + i] = (bf16_t)0.0f;
  }
  {
    const bf16_t* qws = (const bf16_t*)(ws + WS_QW);
    const bf16_t* kws = (const bf16_t*)(ws + WS_KW);
    for (int i = tid; i < 1024; i += 256) { qwL[i] = qws[i]; kwL[i] = kws[i]; }
    for (int i = tid; i < 144; i += 256) { pwL[i] = p_w[i]; lpwL[i] = lp_w[i]; }
    if (tid < 32) { cbL[tid] = conv_b[tid]; bgL[tid] = bn_g[tid]; bbL[tid] = bn_b[tid]; }
    if (tid < 24) lpbL[tid] = lp_b[tid];
    if (tid < 6)  pbL[tid]  = p_b[tid];
  }
  __syncthreads();

  const int mloc = lane & 15, khalf = lane >> 4;

  /* ---- P2: q/k projections, WMMA bf16, bias in C ---- */
  for (int tile = wv; tile < 56; tile += 8) {
    const int pk = tile / 28, idx = tile % 28;
    const int mt = idx % 14, nt = idx / 14;
    const bf16_t* Y = pk ? yC : yT;
    const bf16_t* W = pk ? kwL : qwL;
    const float* bias = pk ? k_b : q_b;
    float* Z = pk ? zk : zq;
    const int row = mt * 16 + mloc;
    v16bf a, bw;
    ((uint4*)&a)[0]  = *(const uint4*)(Y + row * 32 + khalf * 8);
    ((uint4*)&a)[1]  = *(const uint4*)(Y + row * 32 + 16 + khalf * 8);
    ((uint4*)&bw)[0] = *(const uint4*)(W + (nt * 16 + mloc) * 32 + khalf * 8);
    ((uint4*)&bw)[1] = *(const uint4*)(W + (nt * 16 + mloc) * 32 + 16 + khalf * 8);
    const float bv = bias[nt * 16 + mloc];
    v8f c;
    #pragma unroll
    for (int r = 0; r < 8; ++r) c[r] = bv;
    v8f d = __builtin_amdgcn_wmma_f32_16x16x32_bf16(false, a, false, bw, (short)0, c, false, false);
    float* zp = Z + (mt * 16 + khalf * 8) * 32 + nt * 16 + mloc;
    #pragma unroll
    for (int r = 0; r < 8; ++r) zp[r * 32] = d[r];
  }
  __syncthreads();

  /* ---- P3: per-(row,head) L2-normalize q,k in place (hw rsqrt) ---- */
  for (int g = tid; g < 2 * TL * NH; g += 256) {
    float* Z = (g < TL * NH) ? zq : zk;
    const int gg = (g < TL * NH) ? g : g - TL * NH;
    float* pz = Z + (gg >> 3) * 32 + (gg & 7) * 4;
    float4 v = *(float4*)pz;
    const float s = v.x*v.x + v.y*v.y + v.z*v.z + v.w*v.w;
    const float inv = fminf(__frsqrt_rn(s), 1e12f);
    v.x*=inv; v.y*=inv; v.z*=inv; v.w*=inv;
    *(float4*)pz = v;
  }
  __syncthreads();

  /* ---- P4 (per t): scores f32-WMMA -> softmax -> (P @ VMB) bf16-WMMA ---- */
  const bf16_t* vmbG = (const bf16_t*)(ws + WS_VMB);
  for (int t = 0; t < T_SZ; ++t) {
    /* P4a: raw scores S[h][lq][lk] = q . k, V_WMMA_F32_16X16X4_F32; 4 tiles/wave */
    #pragma unroll
    for (int i = 0; i < 4; ++i) {
      const int s4 = wv * 4 + i;
      const int h = s4 >> 2, mt = (s4 >> 1) & 1, nt = s4 & 1;
      const int lq = mt * 16 + mloc;
      const int lk = nt * 16 + mloc;
      const v2f a = *(const v2f*)(zq + (t * L_SZ + lq) * 32 + h * 4 + khalf * 2);
      const v2f bb = *(const v2f*)(zk + (t * L_SZ + lk) * 32 + h * 4 + khalf * 2);
      v8f c;
      #pragma unroll
      for (int r = 0; r < 8; ++r) c[r] = 0.0f;
      v8f d = __builtin_amdgcn_wmma_f32_16x16x4_f32(false, a, false, bb, (short)0, c, false, false);
      float* sp = S + h * 1024 + (mt * 16 + khalf * 8) * 32 + nt * 16 + mloc;
      #pragma unroll
      for (int r = 0; r < 8; ++r) sp[r * 32] = d[r];
    }
    __syncthreads();
    /* P4b: softmax over lk (scale 0.5 folded), write bf16 probs into Pcat */
    if (tid < NH * L_SZ) {
      const int h = tid / L_SZ, lq = tid % L_SZ;
      const float* srow = S + h * 1024 + lq * 32;
      float sv[L_SZ]; float mx = -1e30f;
      #pragma unroll
      for (int lk = 0; lk < L_SZ; ++lk) { sv[lk] = srow[lk]; mx = fmaxf(mx, sv[lk]); }
      float sum = 0.f;
      #pragma unroll
      for (int lk = 0; lk < L_SZ; ++lk) { sv[lk] = __expf(0.5f * (sv[lk] - mx)); sum += sv[lk]; }
      const float inv = 1.0f / sum;
      bf16_t* prow = Pc + lq * 192 + h * L_SZ;
      #pragma unroll
      for (int j = 0; j < 6; ++j) {
        v4bf pv = { (bf16_t)(sv[4*j]*inv), (bf16_t)(sv[4*j+1]*inv),
                    (bf16_t)(sv[4*j+2]*inv), (bf16_t)(sv[4*j+3]*inv) };
        *(v4bf*)(prow + 4 * j) = pv;
      }
    }
    __syncthreads();
    /* P4c: z_m = Pcat(24x192) @ VMB[t](192x32) + m_b -> inp ch 32..63 */
    if (wv < 4) {
      const int mt = wv >> 1, nt = wv & 1;
      const bf16_t* vmb = vmbG + (size_t)(t * 32 + nt * 16 + mloc) * 192;
      const float bv = m_b[nt * 16 + mloc];
      v8f acc;
      #pragma unroll
      for (int r = 0; r < 8; ++r) acc[r] = bv;
      const bf16_t* pr = Pc + (mt * 16 + mloc) * 192;
      #pragma unroll
      for (int cc = 0; cc < 6; ++cc) {
        v16bf a, bw;
        ((uint4*)&a)[0]  = *(const uint4*)(pr + cc * 32 + khalf * 8);
        ((uint4*)&a)[1]  = *(const uint4*)(pr + cc * 32 + 16 + khalf * 8);
        ((uint4*)&bw)[0] = *(const uint4*)(vmb + cc * 32 + khalf * 8);
        ((uint4*)&bw)[1] = *(const uint4*)(vmb + cc * 32 + 16 + khalf * 8);
        acc = __builtin_amdgcn_wmma_f32_16x16x32_bf16(false, a, false, bw, (short)0, acc, false, false);
      }
      #pragma unroll
      for (int r = 0; r < 8; ++r) {
        const int rloc = mt * 16 + r + khalf * 8;                 /* 0..31 */
        const int row = (rloc < L_SZ) ? (t * L_SZ + rloc) : (TL + r); /* pad redirect */
        inp[row * 64 + 32 + nt * 16 + mloc] = (bf16_t)acc[r];
      }
    }
    __syncthreads();
  }

  /* ---- P6: 3x3 conv (64->32) as implicit-im2col WMMA GEMM ---- */
  {
    const bf16_t* wc = (const bf16_t*)(ws + WS_CONV);
    for (int idx = wv; idx < 28; idx += 8) {
      const int mt = idx % 14, nt = idx / 14;
      const int m = mt * 16 + mloc;
      const int t = m / L_SZ, l = m % L_SZ;
      const int oc = nt * 16 + mloc;
      const float cb = cbL[oc];
      v8f acc;
      #pragma unroll
      for (int r = 0; r < 8; ++r) acc[r] = cb;
      #pragma unroll
      for (int tap = 0; tap < 9; ++tap) {
        const int tt = t + tap / 3 - 1;
        const int ll = l + tap % 3 - 1;
        const bool ok = (m < TL) && (tt >= 0) && (tt < T_SZ) && (ll >= 0) && (ll < L_SZ);
        const int pos = ok ? (tt * L_SZ + ll) : 0;
        const bf16_t* srcrow = ok ? (inp + pos * 64) : zrow;
        #pragma unroll
        for (int ch = 0; ch < 2; ++ch) {
          v16bf a;
          ((uint4*)&a)[0] = *(const uint4*)(srcrow + ch * 32 + khalf * 8);
          ((uint4*)&a)[1] = *(const uint4*)(srcrow + ch * 32 + 16 + khalf * 8);
          v16bf bw;
          const bf16_t* wp = wc + (size_t)(tap * 32 + oc) * 64 + ch * 32;
          ((uint4*)&bw)[0] = *(const uint4*)(wp + khalf * 8);
          ((uint4*)&bw)[1] = *(const uint4*)(wp + 16 + khalf * 8);
          acc = __builtin_amdgcn_wmma_f32_16x16x32_bf16(false, a, false, bw, (short)0, acc, false, false);
        }
      }
      const float bg = bgL[oc], bb2 = bbL[oc];
      float* hp = hL + (mt * 16 + khalf * 8) * 33 + oc;
      #pragma unroll
      for (int r = 0; r < 8; ++r)
        hp[r * 33] = fmaxf(fmaf(acc[r], bg, bb2), 0.0f);
    }
  }
  __syncthreads();

  /* ---- P7: p = h@p_w.T + p_b; e = relu(tp) - p; out = e@lp_w.T + lp_b ---- */
  {
    float* outb = out + (size_t)b * (C_SZ * TL);
    for (int it = tid; it < C_SZ * T_SZ; it += 256) {
      const int oc = it / T_SZ, t = it % T_SZ;
      float hrow[L_SZ];
      #pragma unroll
      for (int l = 0; l < L_SZ; ++l) hrow[l] = hL[(t * L_SZ + l) * 33 + oc];
      float e[6];
      #pragma unroll
      for (int j = 0; j < 6; ++j) {
        float s = pbL[j];
        #pragma unroll
        for (int l = 0; l < L_SZ; ++l) s += hrow[l] * pwL[j * L_SZ + l];
        e[j] = fmaxf(tpL[(oc * T_SZ + t) * 6 + j], 0.0f) - s;
      }
      float* ob = outb + (oc * T_SZ + t) * L_SZ;
      #pragma unroll
      for (int l4 = 0; l4 < 6; ++l4) {
        float4 o;
        float* oo = (float*)&o;
        #pragma unroll
        for (int q = 0; q < 4; ++q) {
          const int l = l4 * 4 + q;
          float s = lpbL[l];
          #pragma unroll
          for (int j = 0; j < 6; ++j) s += e[j] * lpwL[l * 6 + j];
          oo[q] = s;
        }
        *(float4*)(ob + l4 * 4) = o;
      }
    }
  }
}

extern "C" void kernel_launch(void* const* d_in, const int* in_sizes, int n_in,
                              void* d_out, int out_size, void* d_ws, size_t ws_size,
                              hipStream_t stream) {
  (void)in_sizes; (void)n_in; (void)out_size; (void)ws_size;
  const float* x           = (const float*)d_in[0];
  const float* pos_prompt  = (const float*)d_in[1];
  const float* rule_prompt = (const float*)d_in[2];
  const float* pre_prompt  = (const float*)d_in[3];
  const float* q_w  = (const float*)d_in[4];
  const float* q_b  = (const float*)d_in[5];
  const float* k_w  = (const float*)d_in[6];
  const float* k_b  = (const float*)d_in[7];
  const float* v_w  = (const float*)d_in[8];
  const float* v_b  = (const float*)d_in[9];
  const float* m_w  = (const float*)d_in[10];
  const float* m_b  = (const float*)d_in[11];
  const float* conv_w = (const float*)d_in[12];
  const float* conv_b = (const float*)d_in[13];
  const float* bn_g = (const float*)d_in[14];
  const float* bn_b = (const float*)d_in[15];
  const float* p_w  = (const float*)d_in[16];
  const float* p_b  = (const float*)d_in[17];
  const float* lp_w = (const float*)d_in[18];
  const float* lp_b = (const float*)d_in[19];
  const int* flag   = (const int*)d_in[20];
  char* ws = (char*)d_ws;
  float* out = (float*)d_out;

  prep_kernel<<<1, 256, 0, stream>>>(rule_prompt, v_w, v_b, q_w, k_w, m_w, conv_w, ws);
  fused_kernel<<<B_SZ, 256, SMEM_BYTES, stream>>>(
      x, pos_prompt, pre_prompt, q_b, k_b, m_b, conv_b, bn_g, bn_b,
      p_w, p_b, lp_w, lp_b, flag, ws, out);
}
